// TimeConv_38620345925799
// MI455X (gfx1250) — matrix operand order repaired and checked by reading, not compiled, not run on hardware
//
#include <hip/hip_runtime.h>
#include <hip/hip_bf16.h>

#define L_ 16
#define P_ 32768
#define K_ 8
#define H_ 128
#define H2_ 64
#define F_ 32

// Use TDM (tensor_load_to_lds) for global->LDS staging of prepacked data.
#ifndef USE_TDM
#if defined(__has_builtin)
#if __has_builtin(__builtin_amdgcn_tensor_load_to_lds)
#define USE_TDM 1
#endif
#endif
#endif
#ifndef USE_TDM
#define USE_TDM 0
#endif

typedef __attribute__((ext_vector_type(16))) __bf16 v16bf;
typedef __attribute__((ext_vector_type(8)))  __bf16 v8bf;
typedef __attribute__((ext_vector_type(4)))  __bf16 v4bf;
typedef __attribute__((ext_vector_type(8)))  float  v8f;
typedef __attribute__((ext_vector_type(4)))  unsigned int u32x4;
typedef __attribute__((ext_vector_type(8)))  int i32x8;
typedef __attribute__((ext_vector_type(4)))  int i32x4;

__device__ __forceinline__ float leaky01(float x) { return x > 0.f ? x : 0.1f * x; }

__device__ __forceinline__ v8f wmma_bf16(v16bf a, v16bf b, v8f c) {
    return __builtin_amdgcn_wmma_f32_16x16x32_bf16(
        false, a, false, b, (short)0, c, false, false);
}

// A fragment (16x32 bf16) from row-major LDS: two contiguous 8-elem runs.
__device__ __forceinline__ v16bf load_a(const __bf16* base, int ld, int mBase, int kBase) {
    int lane = threadIdx.x & 31;
    const __bf16* p = base + (mBase + (lane & 15)) * ld + kBase + ((lane >> 4) << 3);
    v8bf lo = *(const v8bf*)(p);
    v8bf hi = *(const v8bf*)(p + 16);
    return __builtin_shufflevector(lo, hi, 0,1,2,3,4,5,6,7,8,9,10,11,12,13,14,15);
}

// B fragment from fragment-major packed LDS: one 32B vector dereference.
__device__ __forceinline__ v16bf load_b_pk(const __bf16* basePk, int Nt, int kTile, int nTile) {
    int lane = threadIdx.x & 31;
    return *((const v16bf*)basePk + (((kTile * Nt + nTile) << 5) | lane));
}

// Pack row-major f32 (K x N) into fragment-major bf16 (global dst, prepack pass).
// Inverse of load_b_pk: K = kT*32 + ((e>>3)<<4) + ((lane>>4)<<3) + (e&7) ; N = nT*16 + (lane&15).
__device__ __forceinline__ void pack_b(const float* __restrict__ W, __bf16* __restrict__ Bp,
                                       int Kdim, int Ndim) {
    const int Nt = Ndim >> 4;
    for (int i = threadIdx.x; i < Kdim * Ndim; i += 256) {
        int k = i / Ndim, n = i & (Ndim - 1);
        int kT = k >> 5, k32 = k & 31;
        int lane = (n & 15) | (((k32 >> 3) & 1) << 4);
        Bp[((((kT * Nt + (n >> 4)) << 5) | lane) << 4) | ((k32 & 7) | ((k32 >> 4) << 3))] = (__bf16)W[i];
    }
}

// ---------------------------------------------------------------------------
// TDM: 1-D tile copy global -> LDS via Tensor Data Mover (D# per ISA ch.8).
// data_size=8B; optional LDS padding: write 64 dwords, skip 64 dwords
// (pad_interval code 5 = 64 dw, pad_amount code 63 = 64 dw).
// amdgpu-toolchain (clang-23) 6-arg builtin form.
// ---------------------------------------------------------------------------
#if USE_TDM
__device__ __forceinline__ void tdm_copy1d(const void* gsrc, void* lds_dst,
                                           unsigned bytes, int interleave) {
    unsigned long long ga = (unsigned long long)(uintptr_t)gsrc;
    unsigned nelem = bytes >> 3;                       // 8-byte units
    u32x4 g0;
    g0[0] = 1u;                                        // count=1 valid descriptor
    g0[1] = (unsigned)(uintptr_t)lds_dst;              // lds_addr (low 32 of flat)
    g0[2] = (unsigned)ga;                              // global_addr[31:0]
    g0[3] = ((unsigned)(ga >> 32) & 0x1FFFFFFu) | 0x80000000u;  // addr[56:32]|type=2
    i32x8 g1;
    g1[0] = (3 << 16) |                                // data_size = 8B
            (interleave ? ((1 << 20) | (5 << 22) | (63 << 25)) : 0);
    g1[1] = (int)((nelem & 0xFFFFu) << 16);            // tensor_dim0 lo -> bits 63:48
    g1[2] = (int)(((nelem >> 16) & 0xFFFFu) | (1u << 16)); // dim0 hi | tensor_dim1=1
    g1[3] = (int)((nelem & 0xFFFFu) << 16);            // tile_dim0 = nelem (<=65535)
    g1[4] = 0;                                         // tile_dim1/2 unused
    g1[5] = (int)nelem;                                // tensor_dim0_stride lo
    g1[6] = 0;
    g1[7] = 0;
    i32x4 z4 = {0, 0, 0, 0};
    i32x8 z8 = {0, 0, 0, 0, 0, 0, 0, 0};
    __builtin_amdgcn_tensor_load_to_lds(g0, g1, z4, z4, z8, 0);
}
#endif
// Fallback / generic vectorized byte copy (also used when !USE_TDM)
__device__ __forceinline__ void lds_copy16(const __bf16* gsrc, __bf16* dst, int nElem) {
    for (int i = threadIdx.x; i < (nElem >> 3); i += 256)
        *((v8bf*)dst + i) = *((const v8bf*)gsrc + i);
}

// ---------------------------------------------------------------------------
// Prepack: weights -> fragment-major bf16 ; feat -> bf16
// ---------------------------------------------------------------------------
__global__ __launch_bounds__(256) void prepack_w_kernel(
    const float* __restrict__ ne_w1, const float* __restrict__ ne_w2,
    const float* __restrict__ sf_w1, const float* __restrict__ sf_w2,
    const float* __restrict__ ow1,
    __bf16* __restrict__ Wn1, __bf16* __restrict__ Wn2,
    __bf16* __restrict__ Ws1, __bf16* __restrict__ Ws2,
    __bf16* __restrict__ Wout)
{
    pack_b(ne_w1, Wn1, H_,  H2_);
    pack_b(ne_w2, Wn2, H2_, H_);
    pack_b(sf_w1, Ws1, F_,  H2_);
    pack_b(sf_w2, Ws2, H2_, H_);
    pack_b(ow1,  Wout, 2 * H_, H_);
}

__global__ __launch_bounds__(256) void prepack_feat_kernel(
    const float* __restrict__ feat, __bf16* __restrict__ featPk)
{
    int i = (blockIdx.x * 256 + threadIdx.x) * 4;   // 4 elems/thread
    const float* fp = feat + i;
    v4bf v = { (__bf16)fp[0], (__bf16)fp[1], (__bf16)fp[2], (__bf16)fp[3] };
    *(v4bf*)&featPk[i] = v;
}

// ---------------------------------------------------------------------------
// Layer 0 / global MLP: rank-1 input MLP, bf16 output
// ---------------------------------------------------------------------------
__global__ __launch_bounds__(128) void mlp_rank1_kernel(
    const float* __restrict__ x, const float* __restrict__ w1,
    const float* __restrict__ b1, const float* __restrict__ w2,
    const float* __restrict__ b2, __bf16* __restrict__ out)
{
    __shared__ float t[H2_];
    const int p = blockIdx.x;
    const int j = threadIdx.x;
    if (j < H2_) t[j] = leaky01(x[p] * w1[j] + b1[j]);
    __syncthreads();
    float s = b2[j];
#pragma unroll 8
    for (int m = 0; m < H2_; ++m) s += t[m] * w2[m * H_ + j];
    out[p * H_ + j] = (__bf16)s;
}

// ---------------------------------------------------------------------------
// GNN layer: gather-mean (bf16 h, halved L2 traffic) + two MLPs, all WMMA.
// Weight/feat tiles staged by the Tensor Data Mover.
// ---------------------------------------------------------------------------
__global__ __launch_bounds__(256) void layer_kernel(
    const __bf16* __restrict__ h_prev, __bf16* __restrict__ h_cur,
    const int*    __restrict__ pred,   int predBase,
    const __bf16* __restrict__ featPk,          // this layer's bf16 feat slice
    const __bf16* __restrict__ Wn1, const __bf16* __restrict__ Wn2,
    const __bf16* __restrict__ Ws1, const __bf16* __restrict__ Ws2,
    const float* __restrict__ neb1, const float* __restrict__ neb2,
    const float* __restrict__ sfb1, const float* __restrict__ sfb2,
    int doRelu)
{
    __shared__ alignas(32) __bf16 As [128 * H_];
    __shared__ alignas(32) __bf16 Af [128 * F_];
    __shared__ alignas(32) __bf16 Bn1[H_  * H2_];
    __shared__ alignas(32) __bf16 Bn2[H2_ * H_];
    __shared__ alignas(32) __bf16 Bs1[F_  * H2_];
    __shared__ alignas(32) __bf16 Bs2[H2_ * H_];
    __shared__ alignas(32) __bf16 Tn [128 * H2_];
    __shared__ alignas(32) __bf16 Ts [128 * H2_];
    __shared__ int   Idx[128 * K_];
    __shared__ float nb1[H2_], nb2[H_], sb1[H2_], sb2[H_];

    const int tid  = threadIdx.x;
    const int row0 = blockIdx.x * 128;

    // ---- stage packed weights + feat tile via TDM (wave 0 issues) ---------
#if USE_TDM
    if (tid < 32) {
        tdm_copy1d(Wn1, Bn1, H_  * H2_ * 2, 0);
        tdm_copy1d(Wn2, Bn2, H2_ * H_  * 2, 0);
        tdm_copy1d(Ws1, Bs1, F_  * H2_ * 2, 0);
        tdm_copy1d(Ws2, Bs2, H2_ * H_  * 2, 0);
        tdm_copy1d(featPk + (size_t)row0 * F_, Af, 128 * F_ * 2, 0);
        __builtin_amdgcn_s_wait_tensorcnt(0);
    }
#else
    lds_copy16(Wn1, Bn1, H_  * H2_);
    lds_copy16(Wn2, Bn2, H2_ * H_);
    lds_copy16(Ws1, Bs1, F_  * H2_);
    lds_copy16(Ws2, Bs2, H2_ * H_);
    lds_copy16(featPk + (size_t)row0 * F_, Af, 128 * F_);
#endif
    if (tid < H2_) { nb1[tid] = neb1[tid]; sb1[tid] = sfb1[tid]; }
    if (tid < H_)  { nb2[tid] = neb2[tid]; sb2[tid] = sfb2[tid]; }
    for (int i = tid; i < 128 * K_; i += 256) Idx[i] = pred[row0 * K_ + i] - predBase;
    __syncthreads();

    // ---- gather + mean of K=8 predecessor rows (bf16, v8 wide) ------------
    for (int i = tid; i < 128 * (H_ / 8); i += 256) {
        int r = i >> 4, c8 = (i & 15) << 3;
        const int* ip = &Idx[r * K_];
        float s[8] = {0.f, 0.f, 0.f, 0.f, 0.f, 0.f, 0.f, 0.f};
#pragma unroll
        for (int k = 0; k < K_; ++k) {
            v8bf v = *(const v8bf*)&h_prev[ip[k] * H_ + c8];
#pragma unroll
            for (int j = 0; j < 8; ++j) s[j] += (float)v[j];
        }
        v8bf o;
#pragma unroll
        for (int j = 0; j < 8; ++j) o[j] = (__bf16)(s[j] * 0.125f);
        *(v8bf*)&As[r * H_ + c8] = o;
    }
    __syncthreads();

    const int wv   = tid >> 5;
    const int lane = tid & 31;
    const int mOff = (wv << 4);
    const int hi   = (lane >> 4) << 3;
    const int nlo  = lane & 15;

    // ---- GEMM1 ------------------------------------------------------------
    v8f accn[4], accs[4];
#pragma unroll
    for (int n = 0; n < 4; ++n) { accn[n] = (v8f){}; accs[n] = (v8f){}; }
#pragma unroll
    for (int ks = 0; ks < 4; ++ks) {
        v16bf a = load_a(As, H_, mOff, ks * 32);
#pragma unroll
        for (int n = 0; n < 4; ++n)
            accn[n] = wmma_bf16(a, load_b_pk(Bn1, 4, ks, n), accn[n]);
    }
    {
        v16bf a = load_a(Af, F_, mOff, 0);
#pragma unroll
        for (int n = 0; n < 4; ++n)
            accs[n] = wmma_bf16(a, load_b_pk(Bs1, 4, 0, n), accs[n]);
    }
    {
        __bf16* tn = Tn + (mOff + hi) * H2_ + nlo;
        __bf16* ts = Ts + (mOff + hi) * H2_ + nlo;
#pragma unroll
        for (int n = 0; n < 4; ++n)
#pragma unroll
            for (int r = 0; r < 8; ++r) {
                int N = n * 16 + nlo;
                tn[r * H2_ + n * 16] = (__bf16)leaky01(accn[n][r] + nb1[N]);
                ts[r * H2_ + n * 16] = (__bf16)leaky01(accs[n][r] + sb1[N]);
            }
    }
    __syncthreads();

    // ---- GEMM2 (ne + sf accumulated into the same C fragments) ------------
    v8f acc2[8];
#pragma unroll
    for (int n = 0; n < 8; ++n) acc2[n] = (v8f){};
#pragma unroll
    for (int ks = 0; ks < 2; ++ks) {
        v16bf a = load_a(Tn, H2_, mOff, ks * 32);
#pragma unroll
        for (int n = 0; n < 8; ++n)
            acc2[n] = wmma_bf16(a, load_b_pk(Bn2, 8, ks, n), acc2[n]);
    }
#pragma unroll
    for (int ks = 0; ks < 2; ++ks) {
        v16bf a = load_a(Ts, H2_, mOff, ks * 32);
#pragma unroll
        for (int n = 0; n < 8; ++n)
            acc2[n] = wmma_bf16(a, load_b_pk(Bs2, 8, ks, n), acc2[n]);
    }

    // ---- epilogue: biases, relu, bf16 store (base + folded const offsets) -
    {
        __bf16* hp = h_cur + (size_t)(row0 + mOff + hi) * H_ + nlo;
#pragma unroll
        for (int n = 0; n < 8; ++n)
#pragma unroll
            for (int r = 0; r < 8; ++r) {
                int N = n * 16 + nlo;
                float v = acc2[n][r] + nb2[N] + sb2[N];
                if (doRelu) v = fmaxf(v, 0.f);
                hp[r * H_ + n * 16] = (__bf16)v;
            }
    }
}

// ---------------------------------------------------------------------------
// Readout: TDM interleaves h_gnn|h_glob into the 256-wide concat A tile
// (write 64 dwords, pad 64 dwords), then one 128x256x128 WMMA GEMM + dot.
// ---------------------------------------------------------------------------
__global__ __launch_bounds__(256) void out_kernel(
    const __bf16* __restrict__ h_gnn, const __bf16* __restrict__ g,
    const __bf16* __restrict__ Wout,  const float* __restrict__ ob1,
    const float* __restrict__ ow2,    const float* __restrict__ ob2,
    float* __restrict__ out)
{
    __shared__ alignas(32) __bf16 Xa[128 * 256];
    __shared__ alignas(32) __bf16 Bw[256 * H_];
    __shared__ float b1s[H_], w2s[H_];

    const int tid  = threadIdx.x;
    const int row0 = blockIdx.x * 128;

#if USE_TDM
    if (tid < 32) {
        tdm_copy1d(h_gnn + (size_t)row0 * H_, Xa,       128 * H_ * 2, 1);
        tdm_copy1d(g     + (size_t)row0 * H_, Xa + H_,  128 * H_ * 2, 1);
        tdm_copy1d(Wout, Bw, 2 * H_ * H_ * 2, 0);
        __builtin_amdgcn_s_wait_tensorcnt(0);
    }
#else
    for (int i = tid; i < 128 * (H_ / 8); i += 256) {
        int r = i >> 4, c8 = (i & 15) << 3;
        *(v8bf*)&Xa[r * 256 + c8]      = *(const v8bf*)&h_gnn[(row0 + r) * H_ + c8];
        *(v8bf*)&Xa[r * 256 + H_ + c8] = *(const v8bf*)&g[(row0 + r) * H_ + c8];
    }
    lds_copy16(Wout, Bw, 2 * H_ * H_);
#endif
    if (tid < H_) { b1s[tid] = ob1[tid]; w2s[tid] = ow2[tid]; }
    __syncthreads();

    const int wv   = tid >> 5;
    const int lane = tid & 31;
    const int mOff = (wv << 4);

    v8f acc[8];
#pragma unroll
    for (int n = 0; n < 8; ++n) acc[n] = (v8f){};
#pragma unroll
    for (int ks = 0; ks < 8; ++ks) {
        v16bf a = load_a(Xa, 256, mOff, ks * 32);
#pragma unroll
        for (int n = 0; n < 8; ++n)
            acc[n] = wmma_bf16(a, load_b_pk(Bw, 8, ks, n), acc[n]);
    }

    const float bias2 = ob2[0];
#pragma unroll
    for (int r = 0; r < 8; ++r) {
        float part = 0.f;
#pragma unroll
        for (int n = 0; n < 8; ++n) {
            int N = n * 16 + (lane & 15);
            part += leaky01(acc[n][r] + b1s[N]) * w2s[N];
        }
#pragma unroll
        for (int off = 1; off < 16; off <<= 1)
            part += __shfl_xor(part, off, 32);
        if ((lane & 15) == 0) {
            int M = mOff + r + ((lane >> 4) << 3);
            out[row0 + M] = part + bias2;
        }
    }
}

// ---------------------------------------------------------------------------
extern "C" void kernel_launch(void* const* d_in, const int* in_sizes, int n_in,
                              void* d_out, int out_size, void* d_ws, size_t ws_size,
                              hipStream_t stream) {
    const float* delay   = (const float*)d_in[0];
    const float* feat    = (const float*)d_in[1];
    const float* PO_feat = (const float*)d_in[2];
    const int*   pred    = (const int*)  d_in[3];
    const float* pi_w1 = (const float*)d_in[5],  *pi_b1 = (const float*)d_in[6];
    const float* pi_w2 = (const float*)d_in[7],  *pi_b2 = (const float*)d_in[8];
    const float* ne_w1 = (const float*)d_in[9],  *ne_b1 = (const float*)d_in[10];
    const float* ne_w2 = (const float*)d_in[11], *ne_b2 = (const float*)d_in[12];
    const float* sf_w1 = (const float*)d_in[13], *sf_b1 = (const float*)d_in[14];
    const float* sf_w2 = (const float*)d_in[15], *sf_b2 = (const float*)d_in[16];
    const float* gl_w1 = (const float*)d_in[17], *gl_b1 = (const float*)d_in[18];
    const float* gl_w2 = (const float*)d_in[19], *gl_b2 = (const float*)d_in[20];
    const float* ow1   = (const float*)d_in[21], *ob1   = (const float*)d_in[22];
    const float* ow2   = (const float*)d_in[23], *ob2   = (const float*)d_in[24];

    // workspace layout (bf16 h halves gather traffic; weights prepacked)
    char* ws = (char*)d_ws;
    __bf16* hA     = (__bf16*)ws;                                  // 8 MB
    __bf16* hB     = (__bf16*)(ws + (size_t)8  * 1024 * 1024);     // 8 MB
    __bf16* g      = (__bf16*)(ws + (size_t)16 * 1024 * 1024);     // 8 MB
    __bf16* featPk = (__bf16*)(ws + (size_t)24 * 1024 * 1024);     // 32 MB
    __bf16* Wn1    = (__bf16*)(ws + (size_t)56 * 1024 * 1024);
    __bf16* Wn2    = Wn1 + H_ * H2_;
    __bf16* Ws1    = Wn2 + H2_ * H_;
    __bf16* Ws2    = Ws1 + F_ * H2_;
    __bf16* Wout   = Ws2 + H2_ * H_;

    prepack_w_kernel<<<1, 256, 0, stream>>>(ne_w1, ne_w2, sf_w1, sf_w2, ow1,
                                            Wn1, Wn2, Ws1, Ws2, Wout);
    prepack_feat_kernel<<<(L_ * P_ * F_) / 1024, 256, 0, stream>>>(feat, featPk);

    mlp_rank1_kernel<<<P_, 128, 0, stream>>>(delay,   pi_w1, pi_b1, pi_w2, pi_b2, hA);
    mlp_rank1_kernel<<<P_, 128, 0, stream>>>(PO_feat, gl_w1, gl_b1, gl_w2, gl_b2, g);

    __bf16* buf[2] = {hA, hB};
    for (int i = 1; i < L_; ++i) {
        layer_kernel<<<P_ / 128, 256, 0, stream>>>(
            buf[(i - 1) & 1], buf[i & 1],
            pred + (size_t)(i - 1) * P_ * K_, (i - 1) * P_,
            featPk + (size_t)i * P_ * F_,
            Wn1, Wn2, Ws1, Ws2,
            ne_b1, ne_b2, sf_b1, sf_b2,
            (i < L_ - 1) ? 1 : 0);
    }

    out_kernel<<<P_ / 128, 256, 0, stream>>>(buf[(L_ - 1) & 1], g,
                                             Wout, ob1, ow2, ob2, (float*)d_out);
}
